// SubNodeEncoder_27127013441957
// MI455X (gfx1250) — matrix operand order repaired and checked by reading, not compiled, not run on hardware
//
#include <hip/hip_runtime.h>

typedef __attribute__((ext_vector_type(16))) _Float16 v16h;
typedef __attribute__((ext_vector_type(8)))  _Float16 v8h;
typedef __attribute__((ext_vector_type(2)))  _Float16 v2h;
typedef __attribute__((ext_vector_type(8)))  float    v8f;

#define TT    6
#define NP    36
#define DD    64
#define ROWS  216
#define XELEM 3888

// ---------------- LDS layout (bytes), total 51200 ----------------
// Region 1 (0..32768): lifetime-phased
//   pS  f16 [224*64] @0      (28672)   phase B -> C
//   hS  f32 [48*64]  @0      (12288)   post-C -> F
//   gS  f32 [48*64]  @12288  (12288)   phase F -> G
//   wgT f16 [64*64]  @24576  ( 8192)   post-C -> F
// Region 2 (32768..47104):
//   aS  f16 [224*32] @32768  (14336)   phase A -> B
//   ptS f32 [36*64]  @32768  ( 9216)   phase C -> D
//   scS f32 [216]    @41984, twS f32 [216] @42848
//   svS f32 [36]     @43712, swS f32 [36]  @43856, gmS f32 [64] @44000
// Region 3 (47104..51200):
//   cwT f16 [64*32]  @47104  ( 4096)   phase A -> B
#define OFF_PS   0
#define OFF_HS   0
#define OFF_GS   12288
#define OFF_WGT  24576
#define OFF_AS   32768
#define OFF_PT   32768
#define OFF_SC   41984
#define OFF_TW   42848
#define OFF_SV   43712
#define OFF_SW   43856
#define OFF_GM   44000
#define OFF_CWT  47104
#define SMEM_BYTES 51200

__device__ __forceinline__ float waveSum(float v) {
#pragma unroll
  for (int m = 16; m > 0; m >>= 1) v += __shfl_xor(v, m, 32);
  return v;
}

__global__ __launch_bounds__(256) void snec_fused(
    const float* __restrict__ x,  const float* __restrict__ cw,
    const float* __restrict__ cb, const float* __restrict__ wt,
    const float* __restrict__ bt, const float* __restrict__ ws,
    const float* __restrict__ bs, const float* __restrict__ wg,
    const float* __restrict__ bg, const float* __restrict__ wi,
    const float* __restrict__ bi, float* __restrict__ out)
{
  extern __shared__ unsigned char smem[];
  _Float16* pS  = (_Float16*)(smem + OFF_PS);
  float*    hS  = (float*)   (smem + OFF_HS);
  float*    gS  = (float*)   (smem + OFF_GS);
  _Float16* wgT = (_Float16*)(smem + OFF_WGT);
  _Float16* aS  = (_Float16*)(smem + OFF_AS);
  float*    ptS = (float*)   (smem + OFF_PT);
  float*    scS = (float*)   (smem + OFF_SC);
  float*    twS = (float*)   (smem + OFF_TW);
  float*    svS = (float*)   (smem + OFF_SV);
  float*    swS = (float*)   (smem + OFF_SW);
  float*    gmS = (float*)   (smem + OFF_GM);
  _Float16* cwT = (_Float16*)(smem + OFF_CWT);

  const int tid  = threadIdx.x;
  const int lane = tid & 31;
  const int wave = tid >> 5;
  const int b    = blockIdx.x;
  const int nlo  = lane & 15;   // N / M-row within tile
  const int ks   = lane >> 4;   // K-half selector (A: kg, B: ks, D: row+8)

  // ---------- Phase A0: zero-fill padded staging buffers ----------
  {
    const v8h z8 = { (_Float16)0.f, (_Float16)0.f, (_Float16)0.f, (_Float16)0.f,
                     (_Float16)0.f, (_Float16)0.f, (_Float16)0.f, (_Float16)0.f };
    v8h* a16 = (v8h*)aS;          // 896 x 16B
#pragma unroll
    for (int it = 0; it < 4; ++it) {
      int i = tid + it * 256;
      if (i < 896) a16[i] = z8;
    }
    ((v8h*)cwT)[tid] = z8;        // 256 x 16B
  }
  __syncthreads();

  // ---------- Phase A1: coalesced x load -> im2col scatter (packed pairs) ----
  {
    const float4* src = (const float4*)(x + (size_t)b * XELEM);
#pragma unroll
    for (int it = 0; it < 4; ++it) {
      int i = tid + it * 256;
      if (i < XELEM / 4) {
        float4 v = src[i];
        int e0 = i * 4;
#pragma unroll
        for (int p = 0; p < 2; ++p) {
          int e  = e0 + 2 * p;                 // even element (c = 0)
          float va = (p == 0) ? v.x : v.z;
          float vb = (p == 0) ? v.y : v.w;
          int t  = e / 648;  int r1 = e - t * 648;
          int yy = r1 / 36;  int r2 = r1 - yy * 36;
          int xx = r2 >> 1;
          int pi = yy / 3, dy = yy - pi * 3;
          int pj = xx / 3, dx = xx - pj * 3;
          int row = t * 36 + pi * 6 + pj;
          int k   = (dy * 3 + dx) * 2;
          v2h pr = { (_Float16)va, (_Float16)vb };
          *(v2h*)(aS + row * 32 + k) = pr;
        }
      }
    }
    // conv weights -> transposed f16 (cwT[d][k]); pad k>=18 already zero
#pragma unroll
    for (int it = 0; it < 5; ++it) {
      int i = tid + it * 256;
      if (i < 18 * DD) {
        int k = i / DD, d = i - k * DD;
        cwT[d * 32 + k] = (_Float16)cw[i];
      }
    }
  }
  __syncthreads();

  // ---------- Phase B: conv GEMM [224x32]x[32x64] via WMMA ----------
  {
    v16h Bc[4];
#pragma unroll
    for (int nt = 0; nt < 4; ++nt) {
      const v8h* cp = (const v8h*)(cwT + (nt * 16 + nlo) * 32 + ks * 8);
      v8h lo = cp[0], hi = cp[2];
#pragma unroll
      for (int h = 0; h < 8; ++h) { Bc[nt][h] = lo[h]; Bc[nt][h + 8] = hi[h]; }
    }
    const v8f zc = {0.f,0.f,0.f,0.f,0.f,0.f,0.f,0.f};
    for (int tile = wave; tile < 14; tile += 8) {
      int row = tile * 16 + nlo;
      const v8h* ap = (const v8h*)(aS + row * 32 + ks * 8);
      v8h lo = ap[0], hi = ap[2];   // k = 8ks..8ks+7 and 16+8ks..16+8ks+7
      v16h a;
#pragma unroll
      for (int h = 0; h < 8; ++h) { a[h] = lo[h]; a[h + 8] = hi[h]; }
#pragma unroll
      for (int nt = 0; nt < 4; ++nt) {
        v8f acc = __builtin_amdgcn_wmma_f32_16x16x32_f16(
            false, a, false, Bc[nt], (short)0, zc, false, false);
        int d = nt * 16 + nlo;
        float bias = cb[d];
#pragma unroll
        for (int v = 0; v < 8; ++v) {
          int r2 = tile * 16 + v + 8 * ks;   // padded pS: unconditional store
          pS[r2 * DD + d] = (_Float16)fmaxf(acc[v] + bias, 0.f);
        }
      }
    }
  }
  __syncthreads();

  // ---------- Phase C: temporal attention pooling ----------
  {
    float w0 = wt[lane], w1 = wt[lane + 32], btv = bt[0];
    for (int r = wave; r < ROWS; r += 8) {
      float partial = (float)pS[r * DD + lane] * w0 +
                      (float)pS[r * DD + lane + 32] * w1;
      float s = waveSum(partial);
      if (lane == 0) scS[r] = s + btv;
    }
  }
  __syncthreads();
  if (tid < NP) {
    float m = -1e30f;
#pragma unroll
    for (int t = 0; t < TT; ++t) m = fmaxf(m, scS[t * NP + tid]);
    float e[TT], sum = 0.f;
#pragma unroll
    for (int t = 0; t < TT; ++t) { e[t] = __expf(scS[t * NP + tid] - m); sum += e[t]; }
    float inv = 1.f / sum;
#pragma unroll
    for (int t = 0; t < TT; ++t) twS[t * NP + tid] = e[t] * inv;
  }
  __syncthreads();
  for (int i = tid; i < NP * DD; i += 256) {
    int n = i >> 6, d = i & 63;
    float s = 0.f;
#pragma unroll
    for (int t = 0; t < TT; ++t) s += (float)pS[(t * NP + n) * DD + d] * twS[t * NP + n];
    ptS[i] = s;
  }
  __syncthreads();   // pS dead from here; region1 reusable

  // ---------- Phase E0: wg -> transposed f16 LDS, zero hS pad, spatial scores --
  {
#pragma unroll
    for (int it = 0; it < 16; ++it) {       // 4096 elements exactly
      int i = tid + it * 256;
      int k = i >> 6, d = i & 63;
      wgT[d * DD + k] = (_Float16)wg[i];
    }
#pragma unroll
    for (int it = 0; it < 3; ++it)          // hS rows 36..47 -> 0
      hS[NP * DD + tid + it * 256] = 0.f;
    float w0 = ws[lane], w1 = ws[lane + 32], bsv = bs[0];
    for (int r = wave; r < NP; r += 8) {
      float partial = ptS[r * DD + lane] * w0 + ptS[r * DD + lane + 32] * w1;
      float s = waveSum(partial);
      if (lane == 0) svS[r] = s + bsv;
    }
  }
  __syncthreads();
  if (tid < NP) {
    float m = -1e30f;
#pragma unroll
    for (int j = 0; j < NP; ++j) m = fmaxf(m, svS[j]);
    float sum = 0.f;
#pragma unroll
    for (int j = 0; j < NP; ++j) sum += __expf(svS[j] - m);
    swS[tid] = __expf(svS[tid] - m) / sum;
  }
  __syncthreads();
  // h = norm_adj @ (pt * sw): grid adjacency, <=4 neighbors, / deg
  for (int i = tid; i < NP * DD; i += 256) {
    int m = i >> 6, d = i & 63;
    int mi = m / 6, mj = m - mi * 6;
    float s = 0.f; int deg = 0;
    if (mj + 1 < 6) { s += ptS[(m + 1) * DD + d] * swS[m + 1]; ++deg; }
    if (mj > 0)     { s += ptS[(m - 1) * DD + d] * swS[m - 1]; ++deg; }
    if (mi + 1 < 6) { s += ptS[(m + 6) * DD + d] * swS[m + 6]; ++deg; }
    if (mi > 0)     { s += ptS[(m - 6) * DD + d] * swS[m - 6]; ++deg; }
    hS[i] = s / (float)deg;
  }
  __syncthreads();

  // ---------- Phase F: GCN GEMM [48x64]x[64x64] via chained WMMAs ----------
  if (wave < 3) {
    v16h Bg[2][4];
#pragma unroll
    for (int kc = 0; kc < 2; ++kc)
#pragma unroll
      for (int nt = 0; nt < 4; ++nt) {
        const v8h* gp = (const v8h*)(wgT + (nt * 16 + nlo) * DD + kc * 32 + ks * 8);
        v8h lo = gp[0], hi = gp[2];
#pragma unroll
        for (int h = 0; h < 8; ++h) { Bg[kc][nt][h] = lo[h]; Bg[kc][nt][h + 8] = hi[h]; }
      }
    int row = wave * 16 + nlo;
    v16h a0, a1;
#pragma unroll
    for (int kc = 0; kc < 2; ++kc) {
      const float4* hp = (const float4*)(hS + row * DD + kc * 32 + 8 * ks);
      float4 cA = hp[0], cB = hp[1];        // k = 8ks..8ks+7
      const float4* hq = (const float4*)(hS + row * DD + kc * 32 + 16 + 8 * ks);
      float4 cC = hq[0], cD = hq[1];        // k = 16+8ks..
      v16h a;
      a[0]=(_Float16)cA.x; a[1]=(_Float16)cA.y; a[2]=(_Float16)cA.z; a[3]=(_Float16)cA.w;
      a[4]=(_Float16)cB.x; a[5]=(_Float16)cB.y; a[6]=(_Float16)cB.z; a[7]=(_Float16)cB.w;
      a[8]=(_Float16)cC.x; a[9]=(_Float16)cC.y; a[10]=(_Float16)cC.z; a[11]=(_Float16)cC.w;
      a[12]=(_Float16)cD.x; a[13]=(_Float16)cD.y; a[14]=(_Float16)cD.z; a[15]=(_Float16)cD.w;
      if (kc == 0) a0 = a; else a1 = a;
    }
    const v8f zc2 = {0.f,0.f,0.f,0.f,0.f,0.f,0.f,0.f};
#pragma unroll
    for (int nt = 0; nt < 4; ++nt) {
      v8f acc = __builtin_amdgcn_wmma_f32_16x16x32_f16(
          false, a0, false, Bg[0][nt], (short)0, zc2, false, false);
      acc = __builtin_amdgcn_wmma_f32_16x16x32_f16(
          false, a1, false, Bg[1][nt], (short)0, acc, false, false);
      int d = nt * 16 + nlo;
      float bias = bg[d];
#pragma unroll
      for (int v = 0; v < 8; ++v) {          // padded gS: unconditional store
        gS[(wave * 16 + v + 8 * ks) * DD + d] = fmaxf(acc[v] + bias, 0.f);
      }
    }
  }
  __syncthreads();

  // ---------- Phase G: region pooling + inter token -> out[10,64] ----------
  float* outb = out + (size_t)b * (10 * DD);
  for (int i = tid; i < 9 * DD; i += 256) {
    int q = i >> 6, d = i & 63;
    int ri = q / 3, rj = q - ri * 3;
    int m0 = (2 * ri) * 6 + 2 * rj;
    float s = gS[m0 * DD + d] + gS[(m0 + 1) * DD + d] +
              gS[(m0 + 6) * DD + d] + gS[(m0 + 7) * DD + d];
    outb[q * DD + d] = s * 0.25f;
  }
  if (tid < DD) {
    float s = 0.f;
#pragma unroll
    for (int m = 0; m < NP; ++m) s += gS[m * DD + tid];
    gmS[tid] = s * (1.f / 36.f);
  }
  __syncthreads();
  if (tid < DD) {
    float s = bi[tid];
#pragma unroll
    for (int e = 0; e < DD; ++e) s += gmS[e] * wi[e * DD + tid];
    outb[9 * DD + tid] = s;
  }
}

extern "C" void kernel_launch(void* const* d_in, const int* in_sizes, int n_in,
                              void* d_out, int out_size, void* d_ws, size_t ws_size,
                              hipStream_t stream) {
  const float* x  = (const float*)d_in[0];
  const float* cw = (const float*)d_in[1];
  const float* cb = (const float*)d_in[2];
  const float* wt = (const float*)d_in[3];
  const float* bt = (const float*)d_in[4];
  const float* ws = (const float*)d_in[5];
  const float* bs = (const float*)d_in[6];
  const float* wg = (const float*)d_in[7];
  const float* bg = (const float*)d_in[8];
  const float* wi = (const float*)d_in[9];
  const float* bi = (const float*)d_in[10];
  int nb = in_sizes[0] / XELEM;   // 8192
  snec_fused<<<dim3(nb), dim3(256), SMEM_BYTES, stream>>>(
      x, cw, cb, wt, bt, ws, bs, wg, bg, wi, bi, (float*)d_out);
}
